// TensorParallelAttention_45088566673620
// MI455X (gfx1250) — matrix operand order repaired and checked by reading, not compiled, not run on hardware
//
#include <hip/hip_runtime.h>
#include <hip/hip_bf16.h>
#include <stdint.h>

// Problem dims (match reference)
constexpr int kB   = 2;
constexpr int kS   = 2048;
constexpr int kHID = 4096;
constexpr int kH   = 32;
constexpr int kKV  = 8;
constexpr int kD   = 128;
constexpr int kQKV = (kH + 2 * kKV) * kD;     // 6144
constexpr float kSCALE = 0.08838834764831845f; // 1/sqrt(128)
// ln(10000)/64 : inv_freq(d) = exp(-(d&63) * this)
constexpr float kROPE_LOG = 0.14391156831212787f;

typedef __attribute__((ext_vector_type(16))) __bf16 bf16x16;
typedef __attribute__((ext_vector_type(8)))  float  f32x8;
typedef __attribute__((ext_vector_type(4)))  unsigned int u32x4;
typedef __attribute__((ext_vector_type(8)))  int    i32x8;
typedef __attribute__((ext_vector_type(4)))  int    i32x4;

union Frag16 { bf16x16 v; uint4 q[2]; };

static __device__ __forceinline__ uint16_t f32_to_bf16_rne(float f) {
    union { float f; unsigned u; } x; x.f = f;
    unsigned r = x.u + 0x7FFFu + ((x.u >> 16) & 1u);
    return (uint16_t)(r >> 16);
}
static __device__ __forceinline__ float bf16_to_f32(uint16_t h) {
    union { unsigned u; float f; } x; x.u = ((unsigned)h) << 16;
    return x.f;
}
static __device__ __forceinline__ float rowmax16(float v) {
    v = fmaxf(v, __shfl_xor(v, 1)); v = fmaxf(v, __shfl_xor(v, 2));
    v = fmaxf(v, __shfl_xor(v, 4)); v = fmaxf(v, __shfl_xor(v, 8));
    return v;
}
static __device__ __forceinline__ float rowsum16(float v) {
    v += __shfl_xor(v, 1); v += __shfl_xor(v, 2);
    v += __shfl_xor(v, 4); v += __shfl_xor(v, 8);
    return v;
}

// ---------------------------------------------------------------------------
// TDM: async 2-D bf16 tile load (tile_d0 contiguous elems x tile_d1 rows)
// D# per CDNA5 ISA:
//   group0: [1:0]=count=1, [63:32]=lds_addr, [120:64]=global_addr, [127:126]=2
//   group1: [17:16]=data_size(1->2B), [79:48]=tensor_dim0, [111:80]=tensor_dim1,
//           [127:112]=tile_dim0, [143:128]=tile_dim1, [207:160]=dim0 stride
// ---------------------------------------------------------------------------
static __device__ __forceinline__ void tdm_load_2d_bf16(
    unsigned lds_addr, const uint16_t* gptr,
    unsigned tile_d0, unsigned tile_d1,
    unsigned tensor_d0, unsigned tensor_d1,
    unsigned long long stride0_elems)
{
    unsigned long long ga = (unsigned long long)(uintptr_t)gptr;
    u32x4 g0;
    g0[0] = 1u;                                         // count=1 (user D#)
    g0[1] = lds_addr;                                   // LDS byte address
    g0[2] = (unsigned)(ga & 0xffffffffu);               // global addr lo
    g0[3] = (unsigned)((ga >> 32) & 0x1ffffffu) | (2u << 30);   // hi | type=2
    i32x8 g1;
    g1[0] = (int)(1u << 16);                            // data_size = 2 bytes
    g1[1] = (int)((tensor_d0 & 0xffffu) << 16);
    g1[2] = (int)((tensor_d0 >> 16) | ((tensor_d1 & 0xffffu) << 16));
    g1[3] = (int)((tensor_d1 >> 16) | (tile_d0 << 16));
    g1[4] = (int)(tile_d1);                             // tile_dim1; tile_dim2=0
    g1[5] = (int)(stride0_elems & 0xffffffffu);
    g1[6] = (int)((stride0_elems >> 32) & 0xffffu);     // dim1 stride = 0 (2D)
    g1[7] = 0;
    i32x4 z4 = {0, 0, 0, 0};
#if defined(__clang_major__) && (__clang_major__ >= 23)
    i32x8 z8 = {};
    __builtin_amdgcn_tensor_load_to_lds(g0, g1, z4, z4, z8, 0);
#else
    __builtin_amdgcn_tensor_load_to_lds(g0, g1, z4, z4, 0);
#endif
}

// ---------------------------------------------------------------------------
// Elementwise f32 -> bf16 (vectorized), so GEMMs read bf16 once instead of
// re-converting f32 tiles per block (saves VALU and halves tile bandwidth).
// ---------------------------------------------------------------------------
__global__ __launch_bounds__(256)
void cvt_f32_to_bf16(const float4* __restrict__ src, uint2* __restrict__ dst, int n4)
{
    int i = blockIdx.x * 256 + threadIdx.x;
    if (i < n4) {
        float4 v = src[i];
        unsigned lo = (unsigned)f32_to_bf16_rne(v.x) |
                      ((unsigned)f32_to_bf16_rne(v.y) << 16);
        unsigned hi = (unsigned)f32_to_bf16_rne(v.z) |
                      ((unsigned)f32_to_bf16_rne(v.w) << 16);
        dst[i] = make_uint2(lo, hi);
    }
}

// ---------------------------------------------------------------------------
// GEMM: C[M,N] = A[M,K] * W[N,K]^T  (A, W bf16; K contiguous in both)
// Block = 256 threads (8 waves). Block tile 128x128, BK=32.
// LDS tiles are filled by the Tensor Data Mover (wave 0 issues, double
// buffered so the DMA of tile k+1 overlaps WMMA compute on tile k).
// Wave (wm in 0..1, wn in 0..3) owns 64x32 = 4x2 WMMA accumulators.
// ---------------------------------------------------------------------------
template<bool OUT_IS_F32>
__global__ __launch_bounds__(256)
void wmma_gemm_nt(const uint16_t* __restrict__ Aptr, const uint16_t* __restrict__ W,
                  void* __restrict__ Cptr, int M, int N, int K)
{
    __shared__ alignas(16) uint16_t As[2][128][32];
    __shared__ alignas(16) uint16_t Bs[2][128][32];

    const int t    = threadIdx.x;
    const int lane = t & 31;
    const int wid  = t >> 5;
    const int wm   = wid >> 2;        // 0..1 -> 64 rows each
    const int wn   = wid & 3;         // 0..3 -> 32 cols each
    const int hi   = lane >> 4;       // half-wave
    const int ln   = lane & 15;
    const int tileM = blockIdx.y * 128;
    const int tileN = blockIdx.x * 128;

    f32x8 acc[4][2] = {};

    // Prologue: DMA first tiles into buffer 0
    if (wid == 0) {
        tdm_load_2d_bf16((unsigned)(uintptr_t)&As[0][0][0],
                         Aptr + (size_t)tileM * K, 32, 128,
                         (unsigned)K, (unsigned)M, (unsigned long long)K);
        tdm_load_2d_bf16((unsigned)(uintptr_t)&Bs[0][0][0],
                         W + (size_t)tileN * K, 32, 128,
                         (unsigned)K, (unsigned)N, (unsigned long long)K);
        __builtin_amdgcn_s_wait_tensorcnt(0);
    }
    __syncthreads();

    int buf = 0;
    for (int kk = 0; kk < K; kk += 32) {
        const int nbuf = buf ^ 1;
        // Prefetch next tiles via TDM while this tile computes
        if ((kk + 32 < K) && wid == 0) {
            tdm_load_2d_bf16((unsigned)(uintptr_t)&As[nbuf][0][0],
                             Aptr + (size_t)tileM * K + kk + 32, 32, 128,
                             (unsigned)K, (unsigned)M, (unsigned long long)K);
            tdm_load_2d_bf16((unsigned)(uintptr_t)&Bs[nbuf][0][0],
                             W + (size_t)tileN * K + kk + 32, 32, 128,
                             (unsigned)K, (unsigned)N, (unsigned long long)K);
        }

        // ---- build fragments per ISA 16-bit A/B layouts ----
        Frag16 fa[4], fb[2];
        const int kba = hi ? 8 : 0;   // A: K pairs split 0/8 by half-wave
        #pragma unroll
        for (int sm = 0; sm < 4; ++sm) {
            const int r = wm * 64 + sm * 16 + ln;
            fa[sm].q[0] = *(const uint4*)&As[buf][r][kba];
            fa[sm].q[1] = *(const uint4*)&As[buf][r][16 + kba];
        }
        const int kbb = hi ? 8 : 0;   // B: K pairs split 0/8 by half-wave
        #pragma unroll
        for (int sn = 0; sn < 2; ++sn) {
            const int c = wn * 32 + sn * 16 + ln;
            const uint4* p = (const uint4*)&Bs[buf][c][kbb];
            fb[sn].q[0] = p[0];
            fb[sn].q[1] = p[2];
        }
        #pragma unroll
        for (int sm = 0; sm < 4; ++sm)
            #pragma unroll
            for (int sn = 0; sn < 2; ++sn)
                acc[sm][sn] = __builtin_amdgcn_wmma_f32_16x16x32_bf16(
                    false, fa[sm].v, false, fb[sn].v,
                    (short)0, acc[sm][sn], false, false);

        // Drain the prefetch DMA, then publish the next buffer
        if (wid == 0) __builtin_amdgcn_s_wait_tensorcnt(0);
        __syncthreads();
        buf = nbuf;
    }

    // ---- epilogue: C layout lane=N, element r -> M = r + 8*hi ----
    #pragma unroll
    for (int sm = 0; sm < 4; ++sm)
        #pragma unroll
        for (int sn = 0; sn < 2; ++sn)
            #pragma unroll
            for (int r = 0; r < 8; ++r) {
                int grow = tileM + wm * 64 + sm * 16 + r + 8 * hi;
                int gcol = tileN + wn * 32 + sn * 16 + ln;
                if (OUT_IS_F32)
                    ((float*)Cptr)[(size_t)grow * N + gcol] = acc[sm][sn][r];
                else
                    ((uint16_t*)Cptr)[(size_t)grow * N + gcol] =
                        f32_to_bf16_rne(acc[sm][sn][r]);
            }
}

// ---------------------------------------------------------------------------
// RoPE + layout: qkv bf16 [B*S, 6144] ->
//   Qb [B,H,S,D] bf16 (roped), Kb [B,KV,S,D] bf16 (roped),
//   Vt [B,KV,D,S] bf16 (transposed so PV B-fragments load contiguously).
// ---------------------------------------------------------------------------
__global__ __launch_bounds__(256)
void rope_kernel(const uint16_t* __restrict__ qkv, const int* __restrict__ pos_ids,
                 uint16_t* __restrict__ Qb, uint16_t* __restrict__ Kb,
                 uint16_t* __restrict__ Vt)
{
    const int bs = blockIdx.x;          // b*S + s
    const int b  = bs / kS;
    const int s  = bs % kS;
    const float pos = (float)pos_ids[bs];
    const size_t row = (size_t)bs * kQKV;

    for (int idx = threadIdx.x; idx < kQKV; idx += 256) {
        const int d = idx & 127;
        if (idx < (kH + kKV) * kD) {            // q or k: apply RoPE
            float x  = bf16_to_f32(qkv[row + idx]);
            int   fi = d & 63;
            float inv_freq = __expf(-(float)fi * kROPE_LOG);
            float ang = pos * inv_freq;
            float c = __cosf(ang), sn = __sinf(ang);
            int   pidx = (d < 64) ? idx + 64 : idx - 64;
            float xp = bf16_to_f32(qkv[row + pidx]);
            float rot = (d < 64) ? -xp : xp;
            uint16_t y = f32_to_bf16_rne(x * c + rot * sn);
            if (idx < kH * kD) {
                int h = idx >> 7;
                Qb[(((size_t)b * kH + h) * kS + s) * kD + d] = y;
            } else {
                int h = (idx - kH * kD) >> 7;
                Kb[(((size_t)b * kKV + h) * kS + s) * kD + d] = y;
            }
        } else {                                 // v: transpose copy
            int h = (idx - (kH + kKV) * kD) >> 7;
            Vt[(((size_t)b * kKV + h) * kD + d) * kS + s] = qkv[row + idx];
        }
    }
}

// ---------------------------------------------------------------------------
// Flash attention: one wave per (b, h, 16-query tile). Streams 32 keys/iter:
//   S(16x32) = Q(16x128) x K^T       -> 8 x v_wmma_f32_16x16x32_bf16
//   online softmax in f32 (shuffle row-reductions match C layout)
//   P C-layout -> A-layout via LDS bounce (same-wave DS is in-order)
//   O(16x128) += P(16x32) x V        -> 8 x v_wmma_f32_16x16x32_bf16
// ---------------------------------------------------------------------------
__global__ __launch_bounds__(32)
void flash_attn_kernel(const uint16_t* __restrict__ Qb,
                       const uint16_t* __restrict__ Kb,
                       const uint16_t* __restrict__ Vt,
                       uint16_t* __restrict__ Aout)
{
    const int bid = blockIdx.x;
    const int qt  = bid & (kS / 16 - 1);
    const int h   = (bid >> 7) & (kH - 1);
    const int b   = bid >> 12;
    const int kvh = h >> 2;                 // GQA: 4 q-heads per kv head
    const int qbase = qt * 16;

    const int lane = threadIdx.x;
    const int hi   = lane >> 4;
    const int ln   = lane & 15;

    __shared__ alignas(16) uint16_t pt[16][32];

    // Preload Q A-fragments (4 chunks of 16x32 over D)
    Frag16 aq[4];
    {
        const uint16_t* qrow = Qb + (((size_t)b * kH + h) * kS + qbase + ln) * kD;
        const int kba = hi ? 8 : 0;
        #pragma unroll
        for (int c = 0; c < 4; ++c) {
            aq[c].q[0] = *(const uint4*)(qrow + 32 * c + kba);
            aq[c].q[1] = *(const uint4*)(qrow + 32 * c + 16 + kba);
        }
    }
    const uint16_t* Kbase = Kb + ((size_t)b * kKV + kvh) * kS * kD;
    const uint16_t* Vbase = Vt + ((size_t)b * kKV + kvh) * kD * kS;

    f32x8 O[8] = {};
    float mrow[8], lrow[8];
    #pragma unroll
    for (int r = 0; r < 8; ++r) { mrow[r] = -1e30f; lrow[r] = 0.f; }

    const int jend = qbase + 16;            // causal: keys <= qbase+15
    for (int j = 0; j < jend; j += 32) {
        // ---- scores: two 16x16 tiles over 32 keys ----
        f32x8 s0 = {}, s1 = {};
        const int kbb = hi ? 8 : 0;
        #pragma unroll
        for (int c = 0; c < 4; ++c) {
            Frag16 bk0, bk1;
            const uint16_t* p0 = Kbase + (size_t)(j + ln) * kD + 32 * c + kbb;
            const uint16_t* p1 = Kbase + (size_t)(j + 16 + ln) * kD + 32 * c + kbb;
            bk0.q[0] = ((const uint4*)p0)[0]; bk0.q[1] = ((const uint4*)p0)[2];
            bk1.q[0] = ((const uint4*)p1)[0]; bk1.q[1] = ((const uint4*)p1)[2];
            s0 = __builtin_amdgcn_wmma_f32_16x16x32_bf16(false, aq[c].v, false, bk0.v,
                                                         (short)0, s0, false, false);
            s1 = __builtin_amdgcn_wmma_f32_16x16x32_bf16(false, aq[c].v, false, bk1.v,
                                                         (short)0, s1, false, false);
        }

        // ---- online softmax (element r <-> query row r + 8*hi, col = ln) ----
        float corr[8];
        #pragma unroll
        for (int r = 0; r < 8; ++r) {
            const int qi  = qbase + r + 8 * hi;
            const int k0i = j + ln;
            const int k1i = j + 16 + ln;
            float v0 = s0[r] * kSCALE; if (k0i > qi) v0 = -1e30f;
            float v1 = s1[r] * kSCALE; if (k1i > qi) v1 = -1e30f;
            float mx   = rowmax16(fmaxf(v0, v1));
            float mnew = fmaxf(mrow[r], mx);
            float cr   = __expf(mrow[r] - mnew);
            float p0   = __expf(v0 - mnew);
            float p1   = __expf(v1 - mnew);
            lrow[r] = lrow[r] * cr + rowsum16(p0 + p1);
            mrow[r] = mnew;
            corr[r] = cr;
            pt[r + 8 * hi][ln]      = f32_to_bf16_rne(p0);
            pt[r + 8 * hi][16 + ln] = f32_to_bf16_rne(p1);
        }
        // same-wave DS is in-order; fence the compiler + drain DS counter
        asm volatile("s_wait_dscnt 0x0" ::: "memory");

        // P as A-fragment (16x32)
        Frag16 ap;
        const int kba = hi ? 8 : 0;
        ap.q[0] = *(const uint4*)&pt[ln][kba];
        ap.q[1] = *(const uint4*)&pt[ln][16 + kba];

        // rescale accumulators, then O += P x V
        #pragma unroll
        for (int ch = 0; ch < 8; ++ch)
            #pragma unroll
            for (int r = 0; r < 8; ++r)
                O[ch][r] *= corr[r];
        #pragma unroll
        for (int ch = 0; ch < 8; ++ch) {
            Frag16 bv;
            const uint16_t* pv = Vbase + (size_t)(16 * ch + ln) * kS + j + kbb;
            bv.q[0] = ((const uint4*)pv)[0];
            bv.q[1] = ((const uint4*)pv)[2];
            O[ch] = __builtin_amdgcn_wmma_f32_16x16x32_bf16(false, ap.v, false, bv.v,
                                                            (short)0, O[ch], false, false);
        }
    }

    // ---- epilogue: normalize and store bf16 [B,S,H*D] ----
    #pragma unroll
    for (int ch = 0; ch < 8; ++ch)
        #pragma unroll
        for (int r = 0; r < 8; ++r) {
            float o = O[ch][r] / lrow[r];
            int s   = qbase + r + 8 * hi;
            int col = h * kD + 16 * ch + ln;
            Aout[((size_t)b * kS + s) * kHID + col] = f32_to_bf16_rne(o);
        }
}

// ---------------------------------------------------------------------------
extern "C" void kernel_launch(void* const* d_in, const int* in_sizes, int n_in,
                              void* d_out, int out_size, void* d_ws, size_t ws_size,
                              hipStream_t stream)
{
    (void)in_sizes; (void)n_in; (void)out_size; (void)ws_size;
    const float* hidden = (const float*)d_in[0];
    const int*   pos    = (const int*)d_in[1];
    const float* Wqkv   = (const float*)d_in[2];
    const float* Wo     = (const float*)d_in[3];
    float*       out    = (float*)d_out;

    const int M = kB * kS;                         // 4096

    // Workspace layout (all bf16 / uint16):
    uint16_t* hidb  = (uint16_t*)d_ws;                         // [M, 4096]
    uint16_t* wqkvb = hidb  + (size_t)M * kHID;                // [6144, 4096]
    uint16_t* wob   = wqkvb + (size_t)kQKV * kHID;             // [4096, 4096]
    uint16_t* qkv   = wob   + (size_t)kHID * kHID;             // [M, 6144]
    uint16_t* Qb    = qkv   + (size_t)M * kQKV;                // [B,H,S,D]
    uint16_t* Kb    = Qb    + (size_t)kB * kH  * kS * kD;      // [B,KV,S,D]
    uint16_t* Vt    = Kb    + (size_t)kB * kKV * kS * kD;      // [B,KV,D,S]
    uint16_t* Aout  = Vt    + (size_t)kB * kKV * kS * kD;      // [M, 4096]

    // 0) One-shot f32 -> bf16 conversions (halves GEMM tile bandwidth,
    //    removes per-tile conversion VALU from the GEMM hot loop)
    {
        int n4h = (M * kHID) / 4;
        int n4q = (kQKV * kHID) / 4;
        int n4o = (kHID * kHID) / 4;
        cvt_f32_to_bf16<<<dim3((n4h + 255) / 256), 256, 0, stream>>>(
            (const float4*)hidden, (uint2*)hidb, n4h);
        cvt_f32_to_bf16<<<dim3((n4q + 255) / 256), 256, 0, stream>>>(
            (const float4*)Wqkv, (uint2*)wqkvb, n4q);
        cvt_f32_to_bf16<<<dim3((n4o + 255) / 256), 256, 0, stream>>>(
            (const float4*)Wo, (uint2*)wob, n4o);
    }
    // 1) QKV projection (TDM-staged, WMMA): bf16 out
    wmma_gemm_nt<false><<<dim3(kQKV / 128, M / 128), 256, 0, stream>>>(
        hidb, wqkvb, qkv, M, kQKV, kHID);
    // 2) RoPE + Q/K/V layout
    rope_kernel<<<dim3(kB * kS), 256, 0, stream>>>(qkv, pos, Qb, Kb, Vt);
    // 3) Flash attention (causal, GQA)
    flash_attn_kernel<<<dim3(kB * kH * (kS / 16)), 32, 0, stream>>>(Qb, Kb, Vt, Aout);
    // 4) Output projection: f32 out (d_out)
    wmma_gemm_nt<true><<<dim3(kHID / 128, M / 128), 256, 0, stream>>>(
        Aout, wob, out, M, kHID, kHID);
}
